// SelfAttention_32521492365900
// MI455X (gfx1250) — matrix operand order repaired and checked by reading, not compiled, not run on hardware
//
#include <hip/hip_runtime.h>
#include <hip/hip_bf16.h>

#define BATCH 8
#define SEQ   4096
#define DIM   64
// Fold 1/sqrt(64) and log2(e) into Q so scores live in the exp2 domain and
// every exponential is a bare v_exp_f32 (which natively computes 2^x).
#define QSCALE 0.180336880734089f   // 0.125 * log2(e)

typedef __attribute__((ext_vector_type(16))) __bf16 v16bf;
typedef __attribute__((ext_vector_type(8)))  __bf16 v8bf;
typedef __attribute__((ext_vector_type(8)))  float  v8f;

static __device__ __forceinline__ float fexp2(float x) {
    return __builtin_amdgcn_exp2f(x);
}

// Wave-local LDS fence: waves own private LDS regions, so a block barrier is
// unnecessary; same-wave LDS ops are in-order, we only need the data drained
// and the compiler prevented from reordering around the staging buffer.
static __device__ __forceinline__ void wave_lds_fence() {
    __builtin_amdgcn_wave_barrier();
    asm volatile("s_wait_dscnt 0x0" ::: "memory");
    __builtin_amdgcn_wave_barrier();
}

// ---------------------------------------------------------------------------
// Kernel 0: convert fp32 inputs to bf16. Q is pre-scaled by 0.125*log2(e);
// V is transposed to [b][d][k] so WMMA B-fragments of V are contiguous 32B.
// ---------------------------------------------------------------------------
__global__ void attn_convert(const float* __restrict__ Q, const float* __restrict__ K,
                             const float* __restrict__ V,
                             __bf16* __restrict__ Qb, __bf16* __restrict__ Kb,
                             __bf16* __restrict__ Vt) {
    size_t e = (size_t)blockIdx.x * blockDim.x + threadIdx.x;
    if (e >= (size_t)BATCH * SEQ * DIM) return;
    int d = (int)(e % DIM);
    int s = (int)((e / DIM) % SEQ);
    int b = (int)(e / ((size_t)DIM * SEQ));
    Qb[e] = (__bf16)(Q[e] * QSCALE);
    Kb[e] = (__bf16)K[e];
    Vt[((size_t)b * DIM + d) * SEQ + s] = (__bf16)V[e];
}

// ---------------------------------------------------------------------------
// Kernel 1: per-(b,k) column softmax stats over the q axis (exp2 domain).
// One block = one 16-wide k tile; 4 waves stride over q tiles; online max/sum.
// ---------------------------------------------------------------------------
__global__ void attn_stats(const __bf16* __restrict__ Qb, const __bf16* __restrict__ Kb,
                           float* __restrict__ Mout, float* __restrict__ Rout) {
    const int ktile = blockIdx.x % (SEQ / 16);
    const int b     = blockIdx.x / (SEQ / 16);
    const int wave  = threadIdx.x >> 5;
    const int l     = threadIdx.x & 31;
    const int ln    = l & 15;
    const int hi    = (l >> 4) & 1;
    const int base  = hi * 8;
    const int k0    = ktile * 16;

    // B fragments of K^T: lane holds column k=k0+ln, contraction rows d.
    v16bf Bk[2];
    {
        const __bf16* kp = Kb + ((size_t)b * SEQ + k0 + ln) * DIM;
#pragma unroll
        for (int dc = 0; dc < 2; ++dc)
            Bk[dc] = *(const v16bf*)(kp + dc * 32 + hi * 16);
    }

    float mx = -__builtin_inff();
    float sm = 0.0f;

    for (int it = wave; it < SEQ / 16; it += 4) {
        const int q0 = it * 16;
        const __bf16* qp = Qb + ((size_t)b * SEQ + q0 + ln) * DIM;
        v8f c = {};
#pragma unroll
        for (int dc = 0; dc < 2; ++dc) {
            v8bf a0 = *(const v8bf*)(qp + dc * 32 + hi * 16);
            v8bf a1 = *(const v8bf*)(qp + dc * 32 + hi * 16 + 8);
            v16bf a;
#pragma unroll
            for (int i = 0; i < 8; ++i) { a[i] = a0[i]; a[i + 8] = a1[i]; }
            c = __builtin_amdgcn_wmma_f32_16x16x32_bf16(false, a, false, Bk[dc],
                                                        (short)0, c, false, false);
        }
        // lane holds 8 rows of column k0+ln: online max/sum update
        float tm = -__builtin_inff();
#pragma unroll
        for (int r = 0; r < 8; ++r) tm = fmaxf(tm, c[r]);
        float mnew = fmaxf(mx, tm);
        float acc = 0.0f;
#pragma unroll
        for (int r = 0; r < 8; ++r) acc += fexp2(c[r] - mnew);
        sm = sm * fexp2(mx - mnew) + acc;
        mx = mnew;
    }

    // merge lane pair (l, l^16): both hold stats for the same column
    {
        float mo = __shfl_xor(mx, 16, 32);
        float so = __shfl_xor(sm, 16, 32);
        float mn = fmaxf(mx, mo);
        sm = sm * fexp2(mx - mn) + so * fexp2(mo - mn);
        mx = mn;
    }

    __shared__ float smx[4][16];
    __shared__ float ssm[4][16];
    if (l < 16) { smx[wave][ln] = mx; ssm[wave][ln] = sm; }
    __syncthreads();
    if (wave == 0 && l < 16) {
        float m0 = smx[0][ln], s0 = ssm[0][ln];
#pragma unroll
        for (int w = 1; w < 4; ++w) {
            float m1 = smx[w][ln], s1 = ssm[w][ln];
            float mn = fmaxf(m0, m1);
            s0 = s0 * fexp2(m0 - mn) + s1 * fexp2(m1 - mn);
            m0 = mn;
        }
        Mout[(size_t)b * SEQ + k0 + ln] = m0;
        Rout[(size_t)b * SEQ + k0 + ln] = 1.0f / s0;
    }
}

// ---------------------------------------------------------------------------
// Kernel 2: out[q,:] = sum_k exp2(s'[q,k]-m[k])*r[k] * V[k,:].
// Each wave owns a 16-q tile, loops k in tiles of 32. Score WMMAs -> exp2 ->
// bf16 P staged in per-wave LDS (C-layout -> A-layout repack) -> P·V WMMAs.
// No block barriers: LDS regions are wave-private, same-wave DS is in-order.
// ---------------------------------------------------------------------------
__global__ void attn_out(const __bf16* __restrict__ Qb, const __bf16* __restrict__ Kb,
                         const __bf16* __restrict__ Vt, const float* __restrict__ M,
                         const float* __restrict__ R, float* __restrict__ Out) {
    const int qblk = blockIdx.x % (SEQ / 64);
    const int b    = blockIdx.x / (SEQ / 64);
    const int wave = threadIdx.x >> 5;
    const int l    = threadIdx.x & 31;
    const int ln   = l & 15;
    const int hi   = (l >> 4) & 1;
    const int base = hi * 8;
    const int q0   = qblk * 64 + wave * 16;

    __shared__ __align__(16) __bf16 Pbuf[4][16][32];

    // A fragments of Q (reused across the whole k loop)
    v16bf Aq[2];
    {
        const __bf16* qp = Qb + ((size_t)b * SEQ + q0 + ln) * DIM;
#pragma unroll
        for (int dc = 0; dc < 2; ++dc) {
            v8bf a0 = *(const v8bf*)(qp + dc * 32 + hi * 16);
            v8bf a1 = *(const v8bf*)(qp + dc * 32 + hi * 16 + 8);
#pragma unroll
            for (int i = 0; i < 8; ++i) { Aq[dc][i] = a0[i]; Aq[dc][i + 8] = a1[i]; }
        }
    }

    v8f acc[4] = {};  // 16q x 64d output tile as four 16x16 accumulators

    const float* Mb = M + (size_t)b * SEQ;
    const float* Rb = R + (size_t)b * SEQ;

    for (int k0 = 0; k0 < SEQ; k0 += 32) {
        // ---- scores for two 16-wide k tiles, apply exp2, stage bf16 P in LDS
#pragma unroll
        for (int t = 0; t < 2; ++t) {
            const int kcol = k0 + t * 16 + ln;
            const __bf16* kp = Kb + ((size_t)b * SEQ + kcol) * DIM;
            v8f c = {};
#pragma unroll
            for (int dc = 0; dc < 2; ++dc) {
                v16bf bk = *(const v16bf*)(kp + dc * 32 + hi * 16);
                c = __builtin_amdgcn_wmma_f32_16x16x32_bf16(false, Aq[dc], false, bk,
                                                            (short)0, c, false, false);
            }
            const float mk = Mb[kcol];
            const float rk = Rb[kcol];
#pragma unroll
            for (int r = 0; r < 8; ++r) {
                float p = fexp2(c[r] - mk) * rk;
                Pbuf[wave][r + 8 * hi][t * 16 + ln] = (__bf16)p;
            }
        }
        wave_lds_fence();   // drain DS stores; lanes read each other's data next

        // ---- repack P from LDS into A-matrix layout
        v16bf Ap;
        {
            const __bf16* pp = &Pbuf[wave][ln][0];
            v8bf a0 = *(const v8bf*)(pp + hi * 16);
            v8bf a1 = *(const v8bf*)(pp + hi * 16 + 8);
#pragma unroll
            for (int i = 0; i < 8; ++i) { Ap[i] = a0[i]; Ap[i + 8] = a1[i]; }
        }

        // ---- P (16x32) x V (32x64): four 16x16 WMMAs; B-fragments contiguous
#pragma unroll
        for (int dg = 0; dg < 4; ++dg) {
            const __bf16* vp = Vt + ((size_t)b * DIM + dg * 16 + ln) * SEQ + k0 + hi * 16;
            v16bf bv = *(const v16bf*)(vp);
            acc[dg] = __builtin_amdgcn_wmma_f32_16x16x32_bf16(false, Ap, false, bv,
                                                              (short)0, acc[dg], false, false);
        }
        __builtin_amdgcn_wave_barrier();  // keep next iter's DS stores after reads
    }

    // ---- write 16x64 fp32 output tile
#pragma unroll
    for (int dg = 0; dg < 4; ++dg) {
#pragma unroll
        for (int r = 0; r < 8; ++r) {
            int m = r + 8 * hi;
            Out[((size_t)b * SEQ + q0 + m) * DIM + dg * 16 + ln] = acc[dg][r];
        }
    }
}

// ---------------------------------------------------------------------------
extern "C" void kernel_launch(void* const* d_in, const int* in_sizes, int n_in,
                              void* d_out, int out_size, void* d_ws, size_t ws_size,
                              hipStream_t stream) {
    // setup_inputs order: value, key, query
    const float* V = (const float*)d_in[0];
    const float* K = (const float*)d_in[1];
    const float* Q = (const float*)d_in[2];
    float* Out = (float*)d_out;

    char* ws = (char*)d_ws;
    const size_t nElem = (size_t)BATCH * SEQ * DIM;      // 2M elements
    __bf16* Qb = (__bf16*)(ws);
    __bf16* Kb = (__bf16*)(ws + nElem * 2);
    __bf16* Vt = (__bf16*)(ws + nElem * 4);
    float*  Ms = (float*)(ws + nElem * 6);
    float*  Rs = (float*)(ws + nElem * 6 + (size_t)BATCH * SEQ * 4);

    attn_convert<<<dim3((unsigned)((nElem + 255) / 256)), dim3(256), 0, stream>>>(Q, K, V, Qb, Kb, Vt);
    attn_stats<<<dim3(BATCH * (SEQ / 16)), dim3(128), 0, stream>>>(Qb, Kb, Ms, Rs);
    attn_out<<<dim3(BATCH * (SEQ / 64)), dim3(128), 0, stream>>>(Qb, Kb, Vt, Ms, Rs, Out);
}
